// AdvancedTransformer_88759794139600
// MI455X (gfx1250) — matrix operand order repaired and checked by reading, not compiled, not run on hardware
//
#include <hip/hip_runtime.h>

// ---------------------------------------------------------------------------
// MI455X (gfx1250) transformer forward.
// B=8 S=512 DIN=50 D=512 H=8 HD=64 FF=2048 L=6
// All big matmuls: v_wmma_f32_16x16x32_bf16 (fp32 accumulate); every fragment
// feed is ds_load_b128 (LDS tiles stored fragment-major). GEMM register tile
// is 32x64 per wave (2 m-tiles x 4 n-tiles) -> 1.5 ds_load_b128 per WMMA.
// ---------------------------------------------------------------------------

#define Bk   8
#define Sk   512
#define DINk 50
#define Dk   512
#define Hk   8
#define HDk  64
#define FFk  2048
#define Lk   6
#define NTOK 4096        // B*S
#define NREL 257         // 2*MAXREL+1

typedef unsigned short u16;
typedef __attribute__((ext_vector_type(16))) __bf16 v16bf;
typedef __attribute__((ext_vector_type(8)))  float  v8f;
typedef __attribute__((ext_vector_type(8)))  unsigned short us8;  // 16B
typedef __attribute__((ext_vector_type(4)))  unsigned short us4;  // 8B

union Frag16 {
    v16bf v;
    u16   u[16];
    us8   h[2];          // h[0] = elements 0..7, h[1] = 8..15
};

__device__ __forceinline__ u16 f32_to_bf16(float f) {
    unsigned u = __builtin_bit_cast(unsigned, f);
    unsigned r = u + 0x7FFFu + ((u >> 16) & 1u);   // round-to-nearest-even
    return (u16)(r >> 16);
}

// ---------------------------------------------------------------------------
// fp32 -> bf16, 8 elements/thread (b128 in, b128 out). Requires n % 8 == 0.
// ---------------------------------------------------------------------------
__global__ void cvt_bf16_kernel(const float* __restrict__ src,
                                u16* __restrict__ dst, long n) {
    long i = ((long)blockIdx.x * 256 + threadIdx.x) * 8;
    if (i + 8 <= n) {
        float4 f0 = *(const float4*)(src + i);
        float4 f1 = *(const float4*)(src + i + 4);
        us8 o;
        o[0] = f32_to_bf16(f0.x); o[1] = f32_to_bf16(f0.y);
        o[2] = f32_to_bf16(f0.z); o[3] = f32_to_bf16(f0.w);
        o[4] = f32_to_bf16(f1.x); o[5] = f32_to_bf16(f1.y);
        o[6] = f32_to_bf16(f1.z); o[7] = f32_to_bf16(f1.w);
        *(us8*)(dst + i) = o;
    }
}

// ---------------------------------------------------------------------------
// Generic bf16 WMMA GEMM:  C[M,N] = act(A[M,K] @ B[K,N] + bias)
// 256 threads = 8 waves; block tile BM=128 x BN=128; K slabs of 64.
// Wave (wm = w&3, ws = w>>2): rows wm*32..+32 (2 m-tiles), cols ws*64 (4
// n-tiles) -> 8 WMMA per K=32 chunk fed by 4 A + 8 B ds_load_b128.
// LDS: A as [m][k] (stride 72), B transposed as [n][k] (stride 72).
// ---------------------------------------------------------------------------
#define LDA_P 72   // 64 + 8 pad (144B rows, 16B-aligned, bank-staggered)

__global__ __launch_bounds__(256)
void gemm_bf16_kernel(const u16* __restrict__ A, int lda,
                      const u16* __restrict__ Bm, long bsk, long bsn,
                      const float* __restrict__ bias,
                      float* __restrict__ outF, int ldo,
                      u16* __restrict__ outB, int ldob,
                      int M, int N, int K, int relu) {
    __shared__ __align__(16) u16 As[128 * LDA_P];
    __shared__ __align__(16) u16 Bs[128 * LDA_P];
    const int tid  = threadIdx.x;
    const int lane = tid & 31;
    const int wave = tid >> 5;
    const int wm   = wave & 3;       // m-group: rows wm*32 .. wm*32+31
    const int ws   = wave >> 2;      // n-strip: cols ws*64 .. ws*64+63
    const long m0  = (long)blockIdx.x * 128;
    const long n0  = (long)blockIdx.y * 128;
    const int  hw  = lane >> 4;
    const int  nl  = lane & 15;

    v8f acc[2][4] = {};

    for (int k0 = 0; k0 < K; k0 += 64) {
        // ---- stage A slab 128x64: pure b128 copies (4 per thread) ----
        #pragma unroll
        for (int r = 0; r < 4; ++r) {
            int chunk = r * 256 + tid;             // 1024 chunks of 8
            int mi = chunk >> 3, k8 = (chunk & 7) << 3;
            *(us8*)&As[mi * LDA_P + k8] =
                *(const us8*)&A[(m0 + mi) * (long)lda + k0 + k8];
        }
        // ---- stage B slab 128(n) x 64(k), transposed into [n][k] ----
        if (bsn == 1 && n0 + 128 <= N) {
            // weights: coalesced b64 global reads along n, 4 b16 LDS stores
            #pragma unroll
            for (int r = 0; r < 8; ++r) {
                int quad = r * 256 + tid;          // 2048 quads
                int ki = quad >> 5, nq = (quad & 31) << 2;
                us4 vq = *(const us4*)&Bm[(long)(k0 + ki) * bsk + (n0 + nq)];
                #pragma unroll
                for (int j = 0; j < 4; ++j)
                    Bs[(nq + j) * LDA_P + ki] = vq[j];
            }
        } else if (bsk == 1) {
            // rel_emb^T: contiguous along k on both sides -> b128/b128
            #pragma unroll
            for (int r = 0; r < 4; ++r) {
                int chunk = r * 256 + tid;         // 1024 chunks of 8
                int ni = chunk >> 3, k8 = (chunk & 7) << 3;
                long n = n0 + ni;
                us8 v = {};
                if (n < N) v = *(const us8*)&Bm[n * bsn + k0 + k8];
                *(us8*)&Bs[ni * LDA_P + k8] = v;
            }
        } else {
            // generic fallback (unused in this model, kept for safety)
            for (int r = 0; r < 32; ++r) {
                int idx = r * 256 + tid;           // 8192
                int ni = idx >> 6, ki = idx & 63;
                long n = n0 + ni;
                Bs[ni * LDA_P + ki] =
                    (n < N) ? Bm[(long)(k0 + ki) * bsk + n * bsn] : (u16)0;
            }
        }
        // warm next K slab while this one computes
        if (k0 + 64 < K) {
            int mi = tid >> 1;
            __builtin_prefetch(&A[(m0 + mi) * (long)lda + k0 + 64], 0, 1);
            if (bsn == 1)
                __builtin_prefetch(&Bm[(long)(k0 + 64 + (tid >> 3)) * bsk + n0],
                                   0, 1);
        }
        __syncthreads();

        #pragma unroll
        for (int c = 0; c < 2; ++c) {
            Frag16 a0, a1;   // two m-tiles, two b128 loads each
            const u16* ar0 = &As[(wm * 32 + nl) * LDA_P + c * 32 + hw * 8];
            const u16* ar1 = ar0 + 16 * LDA_P;
            a0.h[0] = *(const us8*)ar0;
            a0.h[1] = *(const us8*)(ar0 + 16);
            a1.h[0] = *(const us8*)ar1;
            a1.h[1] = *(const us8*)(ar1 + 16);
            #pragma unroll
            for (int nt = 0; nt < 4; ++nt) {
                Frag16 b;   // two b128 loads, reused by both m-tiles
                const u16* br = &Bs[(ws * 64 + nt * 16 + nl) * LDA_P
                                    + c * 32 + hw * 8];
                b.h[0] = *(const us8*)br;
                b.h[1] = *(const us8*)(br + 16);
                acc[0][nt] = __builtin_amdgcn_wmma_f32_16x16x32_bf16(
                    false, a0.v, false, b.v, (short)0, acc[0][nt], false, false);
                acc[1][nt] = __builtin_amdgcn_wmma_f32_16x16x32_bf16(
                    false, a1.v, false, b.v, (short)0, acc[1][nt], false, false);
            }
        }
        __syncthreads();
    }

    // epilogue: C layout row = j + 8*hw, col = nl (+ tile offsets)
    #pragma unroll
    for (int mt = 0; mt < 2; ++mt) {
        #pragma unroll
        for (int nt = 0; nt < 4; ++nt) {
            long col = n0 + ws * 64 + nt * 16 + nl;
            if (col >= N) continue;
            float bv = bias ? bias[col] : 0.f;
            #pragma unroll
            for (int j = 0; j < 8; ++j) {
                long row = m0 + wm * 32 + mt * 16 + j + 8 * hw;
                float v = acc[mt][nt][j] + bv;
                if (relu) v = fmaxf(v, 0.f);
                if (outF) outF[row * (long)ldo + col] = v;
                if (outB) outB[row * (long)ldob + col] = f32_to_bf16(v);
            }
        }
    }
}

// ---------------------------------------------------------------------------
// Flash attention with precomputed rel bias P[(b,s,h), 257].
// Grid: x = S/64, y = B*H. Block: 128 threads = 4 waves, 16 q rows/wave.
// Keys in blocks of 32. LDS layouts are fragment-major:
//   Ks [t][d] (stride 72)  -> K^T fragments are 2x b128
//   Vs [d][t] (stride 40)  -> V fragments are 2x b128
//   Ps [m][t] (stride 40)  -> probability A-fragments are 2x b128
// ---------------------------------------------------------------------------
#define KS_P 72
#define VS_P 40

__global__ __launch_bounds__(128)
void attn_kernel(const u16* __restrict__ qb, const u16* __restrict__ kb,
                 const u16* __restrict__ vb, const float* __restrict__ P,
                 u16* __restrict__ ctx) {
    __shared__ __align__(16) u16 Ks[32 * KS_P];
    __shared__ __align__(16) u16 Vs[64 * VS_P];
    __shared__ __align__(16) u16 Ps[4][16 * VS_P];

    const int tid  = threadIdx.x;
    const int lane = tid & 31;
    const int wave = tid >> 5;
    const int bh   = blockIdx.y;
    const int b    = bh >> 3;          // / H
    const int hh   = bh & 7;           // % H
    const int s0   = blockIdx.x * 64 + wave * 16;
    const int hw   = lane >> 4;
    const int nl   = lane & 15;

    // q fragments for K=0..63 (two K=32 chunks): 128-bit global loads
    Frag16 aq[2];
    {
        const u16* qrow = qb + ((long)(b * Sk + s0 + nl) * Dk + hh * HDk);
        #pragma unroll
        for (int c = 0; c < 2; ++c) {
            aq[c].h[0] = *(const us8*)(qrow + c * 32 + hw * 8);
            aq[c].h[1] = *(const us8*)(qrow + c * 32 + 16 + hw * 8);
        }
    }

    float mrun[8], lrun[8];
    v8f acc[4] = {};
    #pragma unroll
    for (int j = 0; j < 8; ++j) { mrun[j] = -1e30f; lrun[j] = 0.f; }

    for (int kb0 = 0; kb0 < Sk; kb0 += 32) {
        // ---- stage K block [t][d]: b128/b128 ----
        #pragma unroll
        for (int r = 0; r < 2; ++r) {
            int chunk = r * 128 + tid;             // 256 chunks of 8
            int t = chunk >> 3, d8 = (chunk & 7) << 3;
            *(us8*)&Ks[t * KS_P + d8] =
                *(const us8*)&kb[(long)(b * Sk + kb0 + t) * Dk + hh * HDk + d8];
        }
        // ---- stage V block transposed [d][t]: b64 read + 4 b16 stores ----
        #pragma unroll
        for (int r = 0; r < 4; ++r) {
            int quad = r * 128 + tid;              // 512 quads
            int t = quad >> 4, d = (quad & 15) << 2;
            us4 vq = *(const us4*)&vb[(long)(b * Sk + kb0 + t) * Dk + hh * HDk + d];
            #pragma unroll
            for (int j = 0; j < 4; ++j)
                Vs[(d + j) * VS_P + t] = vq[j];
        }
        __syncthreads();

        // ---- scores 16x32: two n-tiles, K=64 via chained WMMA ----
        v8f sc[2];
        #pragma unroll
        for (int nt = 0; nt < 2; ++nt) {
            v8f s = {};
            #pragma unroll
            for (int c = 0; c < 2; ++c) {
                Frag16 bf;
                const u16* kr = &Ks[(nt * 16 + nl) * KS_P + c * 32 + hw * 8];
                bf.h[0] = *(const us8*)kr;
                bf.h[1] = *(const us8*)(kr + 16);
                s = __builtin_amdgcn_wmma_f32_16x16x32_bf16(
                    false, aq[c].v, false, bf.v, (short)0, s, false, false);
            }
            sc[nt] = s;
        }

        // ---- scale, rel-bias gather, online softmax ----
        #pragma unroll
        for (int j = 0; j < 8; ++j) {
            const int m  = j + 8 * hw;
            const int sg = s0 + m;
            const long prow = ((long)(b * Sk + sg) * Hk + hh) * NREL;

            int t0 = kb0 + nl;
            int d0 = t0 - sg; d0 = d0 < -128 ? -128 : (d0 > 128 ? 128 : d0);
            float v0 = sc[0][j] * 0.125f + P[prow + d0 + 128];
            int t1 = kb0 + 16 + nl;
            int d1 = t1 - sg; d1 = d1 < -128 ? -128 : (d1 > 128 ? 128 : d1);
            float v1 = sc[1][j] * 0.125f + P[prow + d1 + 128];

            float mx = fmaxf(v0, v1);
            #pragma unroll
            for (int o = 1; o < 16; o <<= 1)
                mx = fmaxf(mx, __shfl_xor(mx, o, 32));

            float mnew  = fmaxf(mrun[j], mx);
            float scale = __expf(mrun[j] - mnew);
            mrun[j] = mnew;
            lrun[j] *= scale;
            #pragma unroll
            for (int dt = 0; dt < 4; ++dt) acc[dt][j] *= scale;

            float p0 = __expf(v0 - mnew);
            float p1 = __expf(v1 - mnew);
            float ps = p0 + p1;
            #pragma unroll
            for (int o = 1; o < 16; o <<= 1)
                ps += __shfl_xor(ps, o, 32);
            lrun[j] += ps;

            Ps[wave][m * VS_P + nl]      = f32_to_bf16(p0);
            Ps[wave][m * VS_P + 16 + nl] = f32_to_bf16(p1);
        }
        __syncthreads();

        // ---- ctx += p(16x32) @ v(32x64) ----
        Frag16 ap;
        {
            const u16* pr = &Ps[wave][nl * VS_P + hw * 8];
            ap.h[0] = *(const us8*)pr;
            ap.h[1] = *(const us8*)(pr + 16);
        }
        #pragma unroll
        for (int dt = 0; dt < 4; ++dt) {
            Frag16 bv;
            const u16* vr = &Vs[(dt * 16 + nl) * VS_P + hw * 8];
            bv.h[0] = *(const us8*)vr;
            bv.h[1] = *(const us8*)(vr + 16);
            acc[dt] = __builtin_amdgcn_wmma_f32_16x16x32_bf16(
                false, ap.v, false, bv.v, (short)0, acc[dt], false, false);
        }
        __syncthreads();
    }

    // normalize and write ctx as bf16 (A operand of the O-projection GEMM)
    #pragma unroll
    for (int j = 0; j < 8; ++j) {
        const int m = j + 8 * hw;
        const float inv = 1.f / lrun[j];
        const long rb = (long)(b * Sk + s0 + m) * Dk + hh * HDk;
        #pragma unroll
        for (int dt = 0; dt < 4; ++dt)
            ctx[rb + dt * 16 + nl] = f32_to_bf16(acc[dt][j] * inv);
    }
}

// ---------------------------------------------------------------------------
// LayerNorm over D=512 with optional residual add; writes fp32 + bf16.
// ---------------------------------------------------------------------------
__global__ __launch_bounds__(256)
void ln_kernel(const float* __restrict__ x, const float* __restrict__ res,
               const float* __restrict__ w, const float* __restrict__ bia,
               float* __restrict__ outF, u16* __restrict__ outB) {
    __shared__ float red[256];
    const long row = blockIdx.x;
    const int t = threadIdx.x;
    const float* xr = x + row * Dk;
    float v0 = xr[t]       + (res ? res[row * Dk + t]       : 0.f);
    float v1 = xr[t + 256] + (res ? res[row * Dk + t + 256] : 0.f);

    red[t] = v0 + v1;
    __syncthreads();
    for (int s = 128; s > 0; s >>= 1) {
        if (t < s) red[t] += red[t + s];
        __syncthreads();
    }
    const float mean = red[0] * (1.f / 512.f);
    __syncthreads();

    const float d0 = v0 - mean, d1 = v1 - mean;
    red[t] = d0 * d0 + d1 * d1;
    __syncthreads();
    for (int s = 128; s > 0; s >>= 1) {
        if (t < s) red[t] += red[t + s];
        __syncthreads();
    }
    const float rstd = rsqrtf(red[0] * (1.f / 512.f) + 1e-5f);

    const float y0 = d0 * rstd * w[t]       + bia[t];
    const float y1 = d1 * rstd * w[t + 256] + bia[t + 256];
    outF[row * Dk + t]       = y0;
    outF[row * Dk + t + 256] = y1;
    if (outB) {
        outB[row * Dk + t]       = f32_to_bf16(y0);
        outB[row * Dk + t + 256] = f32_to_bf16(y1);
    }
}

// ---------------------------------------------------------------------------
// Input projection (K=50, too skinny for WMMA) + sinusoidal PE.
// ---------------------------------------------------------------------------
__global__ __launch_bounds__(256)
void input_proj_kernel(const float* __restrict__ x, const float* __restrict__ Win,
                       const float* __restrict__ bin,
                       float* __restrict__ hF, u16* __restrict__ hB) {
    const long idx = (long)blockIdx.x * 256 + threadIdx.x; // over 4096*512
    const int col = idx & (Dk - 1);
    const long row = idx >> 9;
    const int s = (int)(row & (Sk - 1));
    const float* xr = x + row * DINk;
    float a = bin[col];
    #pragma unroll 5
    for (int k = 0; k < DINk; ++k) a += xr[k] * Win[k * Dk + col];
    const int i2 = col & ~1;
    const float div = __expf(-(float)i2 * (9.210340371976184f / 512.f));
    const float ang = (float)s * div;
    a += (col & 1) ? __cosf(ang) : __sinf(ang);
    hF[idx] = a;
    hB[idx] = f32_to_bf16(a);
}

// ---------------------------------------------------------------------------
// Output head: relu(h @ Wp1 + bp1) @ Wp2 + bp2 -> one scalar per token.
// ---------------------------------------------------------------------------
__global__ __launch_bounds__(256)
void head_kernel(const float* __restrict__ h, const float* __restrict__ Wp1,
                 const float* __restrict__ bp1, const float* __restrict__ Wp2,
                 const float* __restrict__ bp2, float* __restrict__ out) {
    __shared__ float hrow[512];
    __shared__ float tr[256];
    const long row = blockIdx.x;
    const int t = threadIdx.x;
    hrow[t]       = h[row * Dk + t];
    hrow[t + 256] = h[row * Dk + t + 256];
    __syncthreads();
    float a = bp1[t];
    for (int k = 0; k < Dk; ++k) a += hrow[k] * Wp1[k * 256 + t];
    a = fmaxf(a, 0.f);
    tr[t] = a * Wp2[t];
    __syncthreads();
    for (int s = 128; s > 0; s >>= 1) {
        if (t < s) tr[t] += tr[t + s];
        __syncthreads();
    }
    if (t == 0) out[row] = tr[0] + bp2[0];
}

// ---------------------------------------------------------------------------
// Host orchestration
// ---------------------------------------------------------------------------
static inline size_t align256(size_t x) { return (x + 255) & ~(size_t)255; }

extern "C" void kernel_launch(void* const* d_in, const int* in_sizes, int n_in,
                              void* d_out, int out_size, void* d_ws, size_t ws_size,
                              hipStream_t stream) {
    (void)in_sizes; (void)n_in; (void)out_size; (void)ws_size;
    const float* x      = (const float*)d_in[0];
    const float* W_in   = (const float*)d_in[1];
    const float* b_in   = (const float*)d_in[2];
    const float* Wq     = (const float*)d_in[3];
    const float* Wkk    = (const float*)d_in[4];
    const float* Wv     = (const float*)d_in[5];
    const float* Wo     = (const float*)d_in[6];
    const float* bo     = (const float*)d_in[7];
    const float* relE   = (const float*)d_in[8];
    const float* W1     = (const float*)d_in[9];
    const float* b1     = (const float*)d_in[10];
    const float* W2     = (const float*)d_in[11];
    const float* b2     = (const float*)d_in[12];
    const float* ln1w   = (const float*)d_in[13];
    const float* ln1b   = (const float*)d_in[14];
    const float* ln2w   = (const float*)d_in[15];
    const float* ln2b   = (const float*)d_in[16];
    const float* olnw   = (const float*)d_in[17];
    const float* olnb   = (const float*)d_in[18];
    const float* Wp1    = (const float*)d_in[19];
    const float* bp1    = (const float*)d_in[20];
    const float* Wp2    = (const float*)d_in[21];
    const float* bp2    = (const float*)d_in[22];
    float* out = (float*)d_out;

    // ---- workspace layout ----
    char* ws = (char*)d_ws;
    size_t off = 0;
    auto carve = [&](size_t bytes) { char* p = ws + off; off += align256(bytes); return p; };
    const size_t nWqkv = (size_t)Lk * Dk * Dk;
    const size_t nWff  = (size_t)Lk * Dk * FFk;
    const size_t nRel  = (size_t)Lk * NREL * HDk;

    u16* wq_bf  = (u16*)carve(nWqkv * 2);
    u16* wk_bf  = (u16*)carve(nWqkv * 2);
    u16* wv_bf  = (u16*)carve(nWqkv * 2);
    u16* wo_bf  = (u16*)carve(nWqkv * 2);
    u16* w1_bf  = (u16*)carve(nWff * 2);
    u16* w2_bf  = (u16*)carve(nWff * 2);
    u16* rel_bf = (u16*)carve(nRel * 2);
    float* h_f  = (float*)carve((size_t)NTOK * Dk * 4);
    u16* h_bf   = (u16*)carve((size_t)NTOK * Dk * 2);
    u16* q_bf   = (u16*)carve((size_t)NTOK * Dk * 2);
    u16* k_bf   = (u16*)carve((size_t)NTOK * Dk * 2);
    u16* v_bf   = (u16*)carve((size_t)NTOK * Dk * 2);
    u16* ctx_bf = (u16*)carve((size_t)NTOK * Dk * 2);
    float* tmp_f = (float*)carve((size_t)NTOK * Dk * 4);
    // P (fp32 [32768,257]) and ff1 (bf16 [4096,2048]) are disjoint in time
    char* big   = carve((size_t)NTOK * Hk * NREL * 4);
    float* P_f  = (float*)big;
    u16* ff1_bf = (u16*)big;

    // ---- weight conversions (idempotent, run every call) ----
    auto cvt = [&](const float* s, u16* d, size_t n) {
        unsigned g = (unsigned)((n / 8 + 255) / 256);
        cvt_bf16_kernel<<<dim3(g), 256, 0, stream>>>(s, d, (long)n);
    };
    cvt(Wq, wq_bf, nWqkv);  cvt(Wkk, wk_bf, nWqkv);
    cvt(Wv, wv_bf, nWqkv);  cvt(Wo, wo_bf, nWqkv);
    cvt(W1, w1_bf, nWff);   cvt(W2, w2_bf, nWff);
    cvt(relE, rel_bf, nRel);

    // ---- input projection + positional encoding ----
    input_proj_kernel<<<dim3(NTOK * Dk / 256), 256, 0, stream>>>(x, W_in, b_in, h_f, h_bf);

    auto gemm = [&](const u16* A, int lda, const u16* Bm, long bsk, long bsn,
                    const float* bias, float* oF, int ldo, u16* oB, int ldob,
                    int M, int N, int K, int relu) {
        dim3 grid(M / 128, (N + 127) / 128);
        gemm_bf16_kernel<<<grid, 256, 0, stream>>>(A, lda, Bm, bsk, bsn, bias,
                                                   oF, ldo, oB, ldob, M, N, K, relu);
    };

    for (int l = 0; l < Lk; ++l) {
        const long wOff = (long)l * Dk * Dk;
        // Q, K, V projections (bf16 outputs only)
        gemm(h_bf, Dk, wq_bf + wOff, Dk, 1, nullptr, nullptr, 0, q_bf, Dk,
             NTOK, Dk, Dk, 0);
        gemm(h_bf, Dk, wk_bf + wOff, Dk, 1, nullptr, nullptr, 0, k_bf, Dk,
             NTOK, Dk, Dk, 0);
        gemm(h_bf, Dk, wv_bf + wOff, Dk, 1, nullptr, nullptr, 0, v_bf, Dk,
             NTOK, Dk, Dk, 0);
        // P = q @ rel_emb^T : A is q reinterpreted [32768,64]; B strides (1,64)
        gemm(q_bf, HDk, rel_bf + (long)l * NREL * HDk, 1, HDk, nullptr,
             P_f, NREL, nullptr, 0, NTOK * Hk, NREL, HDk, 0);
        // attention
        attn_kernel<<<dim3(Sk / 64, Bk * Hk), 128, 0, stream>>>(q_bf, k_bf, v_bf,
                                                                P_f, ctx_bf);
        // O projection + bias -> tmp (fp32)
        gemm(ctx_bf, Dk, wo_bf + wOff, Dk, 1, bo + (long)l * Dk,
             tmp_f, Dk, nullptr, 0, NTOK, Dk, Dk, 0);
        // h = LN(h + attn_out)
        ln_kernel<<<dim3(NTOK), 256, 0, stream>>>(h_f, tmp_f, ln1w + (long)l * Dk,
                                                  ln1b + (long)l * Dk, h_f, h_bf);
        // FFN
        gemm(h_bf, Dk, w1_bf + (long)l * Dk * FFk, FFk, 1, b1 + (long)l * FFk,
             nullptr, 0, ff1_bf, FFk, NTOK, FFk, Dk, 1);
        gemm(ff1_bf, FFk, w2_bf + (long)l * FFk * Dk, Dk, 1, b2 + (long)l * Dk,
             tmp_f, Dk, nullptr, 0, NTOK, Dk, FFk, 0);
        // h = LN(h + ff)
        ln_kernel<<<dim3(NTOK), 256, 0, stream>>>(h_f, tmp_f, ln2w + (long)l * Dk,
                                                  ln2b + (long)l * Dk, h_f, h_bf);
    }

    // final LN (no residual) then MLP head
    ln_kernel<<<dim3(NTOK), 256, 0, stream>>>(h_f, nullptr, olnw, olnb, h_f, h_bf);
    head_kernel<<<dim3(NTOK), 256, 0, stream>>>(h_f, Wp1, bp1, Wp2, bp2, out);
}
